// GSER_88399016887038
// MI455X (gfx1250) — matrix operand order repaired and checked by reading, compiled, not run on hardware
//
#include <hip/hip_runtime.h>

#define BATCH     4096
#define INPUT_DIM 512
#define NACT      2048
#define KBIG      2560   // INPUT_DIM + NACT
#define NGATE     6144

typedef __attribute__((ext_vector_type(16))) __bf16 v16bf;
typedef __attribute__((ext_vector_type(8)))  __bf16 v8bf;
typedef __attribute__((ext_vector_type(8)))  float  v8f;
typedef unsigned short u16;
typedef unsigned int   u32;
typedef unsigned long long u64;

// ---------- bf16 hi/lo split helpers ----------
__device__ __forceinline__ u16 f2bf(float x) {
  u32 u = __float_as_uint(x);
  u32 r = u + 0x7FFFu + ((u >> 16) & 1u);   // RNE
  return (u16)(r >> 16);
}
__device__ __forceinline__ float bf2f(u16 h) {
  return __uint_as_float(((u32)h) << 16);
}
__device__ __forceinline__ void split_bf(float x, u16 &h, u16 &l) {
  h = f2bf(x);
  l = f2bf(x - bf2f(h));
}
__device__ __forceinline__ float sigm(float x) { return 1.0f / (1.0f + __expf(-x)); }

__device__ __forceinline__ v16bf frag16(const u16* p, int off2) {
  v8bf a = *(const v8bf*)p;
  v8bf b = *(const v8bf*)(p + off2);
  return __builtin_shufflevector(a, b, 0,1,2,3,4,5,6,7,8,9,10,11,12,13,14,15);
}
#define WMMA_BF16(A,B,C) __builtin_amdgcn_wmma_f32_16x16x32_bf16(false,(A),false,(B),(short)0,(C),false,false)

// ---------- async global->LDS copy via CDNA5 ISA (GLOBAL_LOAD_ASYNC_TO_LDS_B128) ----------
// VDST = per-lane LDS byte offset, VADDR = per-lane 64-bit global address (GV mode).
__device__ __forceinline__ void cp16(u16* lds, const u16* g) {
  u32 loff = (u32)(u64)(__attribute__((address_space(3))) u16*)lds;
  u64 ga   = (u64)g;
  asm volatile("global_load_async_to_lds_b128 %0, %1, off"
               :: "v"(loff), "v"(ga)
               : "memory");
}
__device__ __forceinline__ void wait_async() {
  asm volatile("s_wait_asynccnt 0x0" ::: "memory");
}

// ---------- conversion kernels ----------
__global__ void cvt_copy_split(const float* __restrict__ src, u16* __restrict__ dh, u16* __restrict__ dl,
                               int rows, int cols, int dstStride, int dstRowOff, int dstColOff) {
  int total = rows * cols;
  for (int i = blockIdx.x * blockDim.x + threadIdx.x; i < total; i += gridDim.x * blockDim.x) {
    int r = i / cols, c = i - r * cols;
    u16 h, l; split_bf(src[i], h, l);
    size_t d = (size_t)(dstRowOff + r) * dstStride + dstColOff + c;
    dh[d] = h; dl[d] = l;
  }
}

// Tiled transpose: dst[srcCol][dstColOff + srcRow] = src[srcRow][srcCol]; dims multiples of 32.
__global__ __launch_bounds__(256)
void cvt_transpose_split(const float* __restrict__ src, u16* __restrict__ dh, u16* __restrict__ dl,
                         int srcCols, int dstStride, int dstColOff) {
  __shared__ float tile[32][33];
  const int bx = blockIdx.x * 32;            // src col base
  const int by = blockIdx.y * 32;            // src row base
  const int tx = threadIdx.x & 31, ty = threadIdx.x >> 5;
#pragma unroll
  for (int j = ty; j < 32; j += 8)
    tile[j][tx] = src[(size_t)(by + j) * srcCols + bx + tx];
  __syncthreads();
#pragma unroll
  for (int j = ty; j < 32; j += 8) {
    float x = tile[tx][j];                    // src[by+tx][bx+j]
    u16 h, l; split_bf(x, h, l);
    size_t d = (size_t)(bx + j) * dstStride + dstColOff + by + tx;
    dh[d] = h; dl[d] = l;
  }
}

// ---------- big GEMM: (4096 x 2560) @ B^T stored N-major (2048 x 2560) ----------
// 64x128 block tile, 8 waves (2 M x 4 N), 32x32 per wave, double-buffered LDS.
// mode 0: store P.  mode 1: semantic-gate + spike epilogue -> out.
__global__ __launch_bounds__(256)
void gemm_big(const u16* __restrict__ Ah, const u16* __restrict__ Al,
              const u16* __restrict__ BTh, const u16* __restrict__ BTl,
              float* __restrict__ Pout,
              const float* __restrict__ state1,
              const float* __restrict__ sg_bias,
              const float* __restrict__ thrp,
              float* __restrict__ out,
              int mode) {
  __shared__ __align__(16) u16 Ash[2][2][64 * 32];    // [buf][plane]
  __shared__ __align__(16) u16 Bsh[2][2][128 * 32];   // [buf][plane], N-major rows

  const int tid  = threadIdx.x;
  const int lane = tid & 31;
  const int wave = tid >> 5;
  const int wm   = wave & 1;
  const int wn   = wave >> 1;
  const int m0   = blockIdx.y * 64;
  const int n0   = blockIdx.x * 128;

  v8f acc[2][2];
#pragma unroll
  for (int i = 0; i < 2; ++i)
#pragma unroll
    for (int j = 0; j < 2; ++j) acc[i][j] = (v8f){};

  const int arow = tid >> 2, acg = (tid & 3) * 8;     // A: 64r x 32k, one 16B chunk/thread/plane
  const int brow = tid >> 1, bcg = (tid & 1) * 16;    // B: 128r x 32k, two 16B chunks/thread/plane

  auto issue = [&](int buf, int k0) {
    cp16(&Ash[buf][0][arow * 32 + acg], Ah + (size_t)(m0 + arow) * KBIG + k0 + acg);
    cp16(&Ash[buf][1][arow * 32 + acg], Al + (size_t)(m0 + arow) * KBIG + k0 + acg);
    size_t gb = (size_t)(n0 + brow) * KBIG + k0 + bcg;
    cp16(&Bsh[buf][0][brow * 32 + bcg],     BTh + gb);
    cp16(&Bsh[buf][0][brow * 32 + bcg + 8], BTh + gb + 8);
    cp16(&Bsh[buf][1][brow * 32 + bcg],     BTl + gb);
    cp16(&Bsh[buf][1][brow * 32 + bcg + 8], BTl + gb + 8);
  };

  const int NIT = KBIG / 32;
  issue(0, 0);
  for (int it = 0; it < NIT; ++it) {
    const int buf = it & 1;
    wait_async();
    __syncthreads();
    if (it + 1 < NIT) issue(buf ^ 1, (it + 1) * 32);

    v16bf afh[2], afl[2], bfh[2], bfl[2];
    const int akk = (lane < 16) ? 0 : 16;
    const int bkk = (lane < 16) ? 0 : 16;
#pragma unroll
    for (int mi = 0; mi < 2; ++mi) {
      int row = wm * 32 + mi * 16 + (lane & 15);
      afh[mi] = frag16(&Ash[buf][0][row * 32 + akk], 8);
      afl[mi] = frag16(&Ash[buf][1][row * 32 + akk], 8);
    }
#pragma unroll
    for (int ni = 0; ni < 2; ++ni) {
      int col = wn * 32 + ni * 16 + (lane & 15);
      bfh[ni] = frag16(&Bsh[buf][0][col * 32 + bkk], 8);
      bfl[ni] = frag16(&Bsh[buf][1][col * 32 + bkk], 8);
    }
#pragma unroll
    for (int mi = 0; mi < 2; ++mi)
#pragma unroll
      for (int ni = 0; ni < 2; ++ni) {
        acc[mi][ni] = WMMA_BF16(afh[mi], bfh[ni], acc[mi][ni]);
        acc[mi][ni] = WMMA_BF16(afh[mi], bfl[ni], acc[mi][ni]);
        acc[mi][ni] = WMMA_BF16(afl[mi], bfh[ni], acc[mi][ni]);
      }
  }

#pragma unroll
  for (int mi = 0; mi < 2; ++mi)
#pragma unroll
    for (int ni = 0; ni < 2; ++ni) {
      int mbase = m0 + wm * 32 + mi * 16 + ((lane >> 4) ? 8 : 0);
      int n = n0 + wn * 32 + ni * 16 + (lane & 15);
#pragma unroll
      for (int r = 0; r < 8; ++r) {
        int m = mbase + r;
        float v = acc[mi][ni][r];
        if (mode == 0) {
          Pout[(size_t)m * NACT + n] = v;
        } else {
          float sg = sigm(v + sg_bias[n]);
          float st = state1[(size_t)m * NACT + n] * sg;
          float tp = thrp[n];
          float thr = (tp > 15.f) ? tp : log1pf(__expf(tp));
          out[(size_t)m * NACT + n] = (st > thr) ? (st - thr) : st;
        }
      }
    }
}

// ---------- gate GEMM (4096 x 512) @ W_gate^T, fused state update ----------
// Block: 64 rows x 32 gate-columns; each wave owns same 16 columns of i/f/o.
__global__ __launch_bounds__(256)
void gemm_gate(const u16* __restrict__ Ah, const u16* __restrict__ Al,
               const u16* __restrict__ BTh, const u16* __restrict__ BTl, // N-major (6144 x 512)
               const float* __restrict__ Pbuf,
               const float* __restrict__ prev,
               const float* __restrict__ leakp,
               float* __restrict__ state1,
               u16* __restrict__ Ahp, u16* __restrict__ Alp) {
  __shared__ __align__(16) u16 Ash[2][2][64 * 32];
  __shared__ __align__(16) u16 Bsh[2][2][96 * 32];

  const int tid  = threadIdx.x;
  const int lane = tid & 31;
  const int wave = tid >> 5;
  const int wm   = wave >> 1;   // 4 waves x 16 rows
  const int wc   = wave & 1;    // 2 column halves
  const int m0   = blockIdx.y * 64;
  const int n0   = blockIdx.x * 32;

  v8f acc[3];
#pragma unroll
  for (int g = 0; g < 3; ++g) acc[g] = (v8f){};

  const int arow = tid >> 2, acg = (tid & 3) * 8;
  const int brow = tid >> 1, bcg = (tid & 1) * 16;    // 96 rows -> tid < 192

  auto issue = [&](int buf, int k0) {
    cp16(&Ash[buf][0][arow * 32 + acg], Ah + (size_t)(m0 + arow) * KBIG + k0 + acg);
    cp16(&Ash[buf][1][arow * 32 + acg], Al + (size_t)(m0 + arow) * KBIG + k0 + acg);
    if (tid < 192) {
      int grow = (brow >> 5) * NACT + n0 + (brow & 31);   // gate*2048 + n0 + col
      size_t gb = (size_t)grow * INPUT_DIM + k0 + bcg;
      cp16(&Bsh[buf][0][brow * 32 + bcg],     BTh + gb);
      cp16(&Bsh[buf][0][brow * 32 + bcg + 8], BTh + gb + 8);
      cp16(&Bsh[buf][1][brow * 32 + bcg],     BTl + gb);
      cp16(&Bsh[buf][1][brow * 32 + bcg + 8], BTl + gb + 8);
    }
  };

  const int NIT = INPUT_DIM / 32;
  issue(0, 0);
  for (int it = 0; it < NIT; ++it) {
    const int buf = it & 1;
    wait_async();
    __syncthreads();
    if (it + 1 < NIT) issue(buf ^ 1, (it + 1) * 32);

    const int akk = (lane < 16) ? 0 : 16;
    const int bkk = (lane < 16) ? 0 : 16;
    int row = wm * 16 + (lane & 15);
    v16bf afh = frag16(&Ash[buf][0][row * 32 + akk], 8);
    v16bf afl = frag16(&Ash[buf][1][row * 32 + akk], 8);
#pragma unroll
    for (int g = 0; g < 3; ++g) {
      int col = g * 32 + wc * 16 + (lane & 15);
      v16bf bfh = frag16(&Bsh[buf][0][col * 32 + bkk], 8);
      v16bf bfl = frag16(&Bsh[buf][1][col * 32 + bkk], 8);
      acc[g] = WMMA_BF16(afh, bfh, acc[g]);
      acc[g] = WMMA_BF16(afh, bfl, acc[g]);
      acc[g] = WMMA_BF16(afl, bfh, acc[g]);
    }
  }

  int mbase = m0 + wm * 16 + ((lane >> 4) ? 8 : 0);
  int n = n0 + wc * 16 + (lane & 15);
  float leak = sigm(leakp[n]);
#pragma unroll
  for (int r = 0; r < 8; ++r) {
    int m = mbase + r;
    float gi = sigm(acc[0][r]);
    float gf = sigm(acc[1][r]);
    float go = sigm(acc[2][r]);
    float P  = Pbuf[(size_t)m * NACT + n];
    float pv = prev[(size_t)m * NACT + n];
    float s  = (1.0f - leak) * (gf * pv) + leak * tanhf(gi * P);
    s *= go;
    state1[(size_t)m * NACT + n] = s;
    u16 h, l; split_bf(s, h, l);
    Ahp[(size_t)m * KBIG + INPUT_DIM + n] = h;
    Alp[(size_t)m * KBIG + INPUT_DIM + n] = l;
  }
}

// ---------- launch ----------
extern "C" void kernel_launch(void* const* d_in, const int* in_sizes, int n_in,
                              void* d_out, int out_size, void* d_ws, size_t ws_size,
                              hipStream_t stream) {
  const float* inputs = (const float*)d_in[0];
  const float* prev   = (const float*)d_in[1];
  const float* W_res  = (const float*)d_in[2];
  const float* W_in   = (const float*)d_in[3];
  const float* W_gate = (const float*)d_in[4];
  const float* leakp  = (const float*)d_in[5];
  const float* thrp   = (const float*)d_in[6];
  const float* sg_k   = (const float*)d_in[7];
  const float* sg_b   = (const float*)d_in[8];
  float* out = (float*)d_out;

  char* ws = (char*)d_ws;
  size_t off = 0;
  auto grab = [&](size_t bytes) -> void* {
    void* p = ws + off;
    off = (off + bytes + 255) & ~(size_t)255;
    return p;
  };
  u16* Ahp = (u16*)grab((size_t)BATCH * KBIG * 2);        // [inputs | prev/state] hi
  u16* Alp = (u16*)grab((size_t)BATCH * KBIG * 2);        // lo
  u16* B1h = (u16*)grab((size_t)NACT * KBIG * 2);         // B1^T: [W_in | W_res^T] N-major
  u16* B1l = (u16*)grab((size_t)NACT * KBIG * 2);
  u16* B2h = (u16*)grab((size_t)NGATE * INPUT_DIM * 2);   // B2^T = W_gate (row copy)
  u16* B2l = (u16*)grab((size_t)NGATE * INPUT_DIM * 2);
  u16* B3h = (u16*)grab((size_t)NACT * KBIG * 2);         // B3^T = sg_kernel^T
  u16* B3l = (u16*)grab((size_t)NACT * KBIG * 2);
  float* P      = (float*)grab((size_t)BATCH * NACT * 4);
  float* state1 = (float*)grab((size_t)BATCH * NACT * 4);
  (void)ws_size; (void)in_sizes; (void)n_in; (void)out_size;

  auto blocks = [](long total) { int b = (int)((total + 1023) / 1024); return b < 1 ? 1 : b; };

  // A planes
  cvt_copy_split<<<blocks((long)BATCH * INPUT_DIM), 256, 0, stream>>>(
      inputs, Ahp, Alp, BATCH, INPUT_DIM, KBIG, 0, 0);
  cvt_copy_split<<<blocks((long)BATCH * NACT), 256, 0, stream>>>(
      prev, Ahp, Alp, BATCH, NACT, KBIG, 0, INPUT_DIM);
  // B1^T: W_in rows copy (k<512), W_res transpose into cols 512..
  cvt_copy_split<<<blocks((long)NACT * INPUT_DIM), 256, 0, stream>>>(
      W_in, B1h, B1l, NACT, INPUT_DIM, KBIG, 0, 0);
  cvt_transpose_split<<<dim3(NACT / 32, NACT / 32), 256, 0, stream>>>(
      W_res, B1h, B1l, NACT, KBIG, INPUT_DIM);
  // B2^T: W_gate straight copy (N-major already)
  cvt_copy_split<<<blocks((long)NGATE * INPUT_DIM), 256, 0, stream>>>(
      W_gate, B2h, B2l, NGATE, INPUT_DIM, INPUT_DIM, 0, 0);
  // B3^T: sg_kernel transpose
  cvt_transpose_split<<<dim3(NACT / 32, KBIG / 32), 256, 0, stream>>>(
      sg_k, B3h, B3l, NACT, KBIG, 0);

  dim3 gridBig(NACT / 128, BATCH / 64);
  gemm_big<<<gridBig, 256, 0, stream>>>(Ahp, Alp, B1h, B1l, P,
                                        nullptr, nullptr, nullptr, nullptr, 0);

  dim3 gridGate(NACT / 32, BATCH / 64);
  gemm_gate<<<gridGate, 256, 0, stream>>>(Ahp, Alp, B2h, B2l, P, prev, leakp,
                                          state1, Ahp, Alp);

  gemm_big<<<gridBig, 256, 0, stream>>>(Ahp, Alp, B3h, B3l, nullptr,
                                        state1, sg_b, thrp, out, 1);
}
